// GPT2SparseMLP_50680614093121
// MI455X (gfx1250) — matrix operand order repaired and checked by reading, not compiled, not run on hardware
//
#include <hip/hip_runtime.h>

typedef __attribute__((ext_vector_type(8)))  float        v8f;
typedef __attribute__((ext_vector_type(16))) __bf16       v16bf;
typedef __attribute__((ext_vector_type(4)))  unsigned int u32x4;
typedef __attribute__((ext_vector_type(8)))  unsigned int u32x8;
typedef __attribute__((ext_vector_type(4)))  float        f32x4;

constexpr int kB = 4, kS = 2048, kH = 1024, kE = 8, kI = 4096, kC = 512;
constexpr int kRows = kB * kE * kC;   // 16384 expert-buffer rows
constexpr int kTok  = kB * kS;        // 8192 tokens

__device__ __forceinline__ unsigned short f2bf(float x) {
  unsigned int u = __float_as_uint(x);
  u += 0x7FFFu + ((u >> 16) & 1u);     // round-to-nearest-even
  return (unsigned short)(u >> 16);
}

__device__ __forceinline__ float gelu_new(float x) {
  float x3 = x * x * x;
  return 0.5f * x * (1.0f + tanhf(0.7978845608028654f * (x + 0.044715f * x3)));
}

// TDM: DMA a [32 x 128] bf16 tile (row stride Kelems) into LDS, padding each
// 64B row with 32B -> matches the PAD=48 LDS layout. Uniform-arg, wave-level.
__device__ __forceinline__ void tdm_load_tile_bf16(
    unsigned long long ga, unsigned lds_addr, int Kelems) {
  const u32x4 gz = (u32x4){0u, 0u, 0u, 0u};
  u32x4 g0;
  g0[0] = 1u;                                       // count=1 (valid user D#)
  g0[1] = lds_addr;                                 // lds_addr (bytes)
  g0[2] = (unsigned)(ga & 0xFFFFFFFFull);           // global_addr[31:0]
  g0[3] = (unsigned)((ga >> 32) & 0x1FFFFFFull)     // global_addr[56:32]
        | (2u << 30);                               // type = 2 ("image")
  u32x8 g1;
  g1[0] = (1u << 16)                                // data_size = 2 bytes
        | (1u << 20)                                // pad_enable
        | (3u << 22)                                // pad_interval = 16 DWORDs
        | (7u << 25);                               // pad_amount   = 8 DWORDs
  g1[1] = ((unsigned)Kelems & 0xFFFFu) << 16;       // tensor_dim0 lo16
  g1[2] = (((unsigned)Kelems >> 16) & 0xFFFFu)      // tensor_dim0 hi16
        | (128u << 16);                             // tensor_dim1 lo16 = 128 rows
  g1[3] = (32u << 16);                              // tile_dim0 = 32 (K chunk)
  g1[4] = 128u;                                     // tile_dim1 = 128 (dim2 = 0)
  g1[5] = (unsigned)Kelems;                         // tensor_dim0_stride lo32
  g1[6] = 0u;
  g1[7] = 0u;
  asm volatile("tensor_load_to_lds %0, %1, %2, %3"
               : : "s"(g0), "s"(g1), "s"(gz), "s"(gz)
               : "memory");
}

// ---------------- Router: one wave per token, top-1 softmax ----------------
__global__ __launch_bounds__(256) void router_kernel(
    const float* __restrict__ x, const float* __restrict__ Wr,
    const float* __restrict__ br, int* __restrict__ eidx, float* __restrict__ mp) {
  const int wave = threadIdx.x >> 5, lane = threadIdx.x & 31;
  const int t = blockIdx.x * 8 + wave;
  const float* xr = x + (size_t)t * kH;
  float acc[kE];
#pragma unroll
  for (int e = 0; e < kE; ++e) acc[e] = 0.0f;
  for (int j = lane; j < kH; j += 32) {
    float xv = xr[j];
    const f32x4* w = (const f32x4*)(Wr + (size_t)j * kE);
    f32x4 w0 = w[0], w1 = w[1];
#pragma unroll
    for (int e = 0; e < 4; ++e) { acc[e] += xv * w0[e]; acc[e + 4] += xv * w1[e]; }
  }
#pragma unroll
  for (int e = 0; e < kE; ++e)
    for (int o = 16; o > 0; o >>= 1) acc[e] += __shfl_xor(acc[e], o, 32);
  if (lane == 0) {
    float logit[kE];
    float best = -1e30f; int bi = 0;
#pragma unroll
    for (int e = 0; e < kE; ++e) {
      logit[e] = acc[e] + br[e];
      if (logit[e] > best) { best = logit[e]; bi = e; }   // first-max like argmax
    }
    float den = 0.0f;
#pragma unroll
    for (int e = 0; e < kE; ++e) den += __expf(logit[e] - best);
    eidx[t] = bi;
    mp[t]   = 1.0f / den;   // softmax max prob
  }
}

__global__ void init_slotmap(int* __restrict__ slot_map) {
  int i = blockIdx.x * blockDim.x + threadIdx.x;
  if (i < kRows) slot_map[i] = -1;
}

// ------- Capacity assignment: per-batch wave32 ballot-based cumsum ---------
__global__ __launch_bounds__(32) void priority_kernel(
    const int* __restrict__ eidx, int* __restrict__ slot_map) {
  const int b = blockIdx.x, lane = threadIdx.x;
  const unsigned int lt = (1u << lane) - 1u;
  int counts[kE];
#pragma unroll
  for (int e = 0; e < kE; ++e) counts[e] = 0;
  for (int s0 = 0; s0 < kS; s0 += 32) {
    const int s = s0 + lane;
    const int e = eidx[b * kS + s];
    int pos = 0;
#pragma unroll
    for (int ee = 0; ee < kE; ++ee) {
      unsigned int m = (unsigned int)__ballot(e == ee);
      if (e == ee) pos = counts[ee] + __popc(m & lt);
      counts[ee] += __popc(m);
    }
    if (pos < kC) slot_map[((size_t)b * kE + e) * kC + pos] = b * kS + s;
  }
}

// -------- Dispatch: gather kept tokens into bf16 [kRows, kH] buffer --------
__global__ __launch_bounds__(128) void dispatch_kernel(
    const float* __restrict__ x, const int* __restrict__ slot_map,
    unsigned short* __restrict__ buf) {
  const int row = blockIdx.x;
  const int t = slot_map[row];
  const int base = threadIdx.x * 8;
  u32x4 o;
  if (t >= 0) {
    const f32x4* src = (const f32x4*)(x + (size_t)t * kH + base);
    f32x4 a = src[0], c = src[1];
    o[0] = (unsigned int)f2bf(a[0]) | ((unsigned int)f2bf(a[1]) << 16);
    o[1] = (unsigned int)f2bf(a[2]) | ((unsigned int)f2bf(a[3]) << 16);
    o[2] = (unsigned int)f2bf(c[0]) | ((unsigned int)f2bf(c[1]) << 16);
    o[3] = (unsigned int)f2bf(c[2]) | ((unsigned int)f2bf(c[3]) << 16);
  } else {
    o = (u32x4){0u, 0u, 0u, 0u};
  }
  *(u32x4*)(buf + (size_t)row * kH + base) = o;
}

// -------- Fallback: out = max_prob * x for every token (dropped path) ------
__global__ __launch_bounds__(256) void fallback_kernel(
    const float* __restrict__ x, const float* __restrict__ mp,
    float* __restrict__ out) {
  const int t = blockIdx.x;
  const float p = mp[t];
  const f32x4* src = (const f32x4*)(x + (size_t)t * kH);
  f32x4* dst = (f32x4*)(out + (size_t)t * kH);
  f32x4 v = src[threadIdx.x];
  v[0] *= p; v[1] *= p; v[2] *= p; v[3] *= p;
  dst[threadIdx.x] = v;
}

// ---- One-shot weight convert+transpose: f32 [E][K][N] -> bf16 [E][N][K] ----
// LDS-tiled so both global reads and writes are coalesced.
template <int K, int N>
__global__ __launch_bounds__(256) void convert_transpose_kernel(
    const float* __restrict__ W, unsigned short* __restrict__ Wt) {
  __shared__ unsigned short T[64][65];
  const int e  = blockIdx.z;
  const int k0 = blockIdx.x * 64;
  const int n0 = blockIdx.y * 64;
  const float* src = W + (size_t)e * K * N;
  unsigned short* dst = Wt + (size_t)e * N * K;
  const int tid = threadIdx.x;
#pragma unroll
  for (int i = 0; i < 16; ++i) {
    int lin = tid + i * 256;
    int r = lin >> 6, c = lin & 63;                       // coalesced along n
    T[c][r] = f2bf(src[(size_t)(k0 + r) * N + (n0 + c)]);
  }
  __syncthreads();
#pragma unroll
  for (int i = 0; i < 16; ++i) {
    int lin = tid + i * 256;
    int r = lin >> 6, c = lin & 63;                       // coalesced along k
    dst[(size_t)(n0 + r) * K + (k0 + c)] = T[r][c];
  }
}

// ----------- Tiled bf16 WMMA GEMM: dual-TDM + ping-pong pipeline -----------
// Both A ([rows][K] bf16) and B (pre-transposed Wt [N][K] bf16) tiles are
// identical strided [32 x 128] bf16 tiles -> same TDM descriptor DMAs both.
// Inner loop: 2 TDM issues + 1 tensorcnt wait + 1 barrier + fragment
// ds_loads + 8 WMMAs. No global loads, no LDS stores, no conversion VALU.
// MODE 0: act = gelu_new(A @ W + bias)           -> bf16 out
// MODE 1: out[slot_map[row]] = mp * (A @ W + b)  -> f32 scatter (combine)
template <int N, int K, int MODE>
__global__ __launch_bounds__(256) void gemm_kernel(
    const unsigned short* __restrict__ Abf,   // [kRows][K] bf16
    const unsigned short* __restrict__ Wt,    // [kE][N][K] bf16 (pre-transposed)
    const float* __restrict__ bias,           // [kE][N]
    unsigned short* __restrict__ actOut,      // MODE 0
    float* __restrict__ dOut,                 // MODE 1
    const int* __restrict__ slot_map,         // MODE 1
    const float* __restrict__ mp) {           // MODE 1
  constexpr int BM = 128, BN = 128, BK = 32, PAD = 48;  // PAD keeps 16B-aligned rows
  constexpr int NIT = K / BK;
  __shared__ unsigned short As[2][BM * PAD];
  __shared__ unsigned short Bs[2][BN * PAD];  // Bs[buf][n][k]
  const int rowBase = blockIdx.y * BM;
  const int colBase = blockIdx.x * BN;
  const int e = (rowBase / kC) % kE;          // C=512 % BM == 0 -> constant per tile
  const int tid  = threadIdx.x;
  const int lane = tid & 31, wave = tid >> 5;
  const int waveR = wave >> 2, waveC = wave & 3;   // 2x4 wave grid -> 64x32 per wave
  const int hh = lane >> 4, l16 = lane & 15;

  const v8f vzero = {0.f, 0.f, 0.f, 0.f, 0.f, 0.f, 0.f, 0.f};
  v8f acc[4][2];
#pragma unroll
  for (int m = 0; m < 4; ++m)
#pragma unroll
    for (int n = 0; n < 2; ++n) acc[m][n] = vzero;

  const unsigned lds_a0 = (unsigned)(uintptr_t)(&As[0][0]);
  const unsigned lds_a1 = (unsigned)(uintptr_t)(&As[1][0]);
  const unsigned lds_b0 = (unsigned)(uintptr_t)(&Bs[0][0]);
  const unsigned lds_b1 = (unsigned)(uintptr_t)(&Bs[1][0]);
  const unsigned long long gaA =
      (unsigned long long)(uintptr_t)(Abf + (size_t)rowBase * K);
  const unsigned long long gaB =
      (unsigned long long)(uintptr_t)(Wt + ((size_t)e * N + colBase) * K);

  // ---- prologue: TDM tile pair 0 -> As[0], Bs[0] ----
  if (tid < 32) {
    tdm_load_tile_bf16(gaA, lds_a0, K);
    tdm_load_tile_bf16(gaB, lds_b0, K);
  }

  for (int it = 0; it < NIT; ++it) {
    const int cur = it & 1;
    // TDM pair(it), issued one iteration ago, has landed in buffers `cur`
    if (tid < 32) __builtin_amdgcn_s_wait_tensorcnt(0);
    // single barrier: all waves' fragment ds_loads of buffers `cur^1` drained
    __syncthreads();
    // issue TDM pair(it+1) -> buffers cur^1 (safe: compute(it-1) done by now)
    if (tid < 32 && it + 1 < NIT) {
      const unsigned long long koff = (unsigned long long)((it + 1) * BK) * 2ull;
      tdm_load_tile_bf16(gaA + koff, cur ? lds_a0 : lds_a1, K);
      tdm_load_tile_bf16(gaB + koff, cur ? lds_b0 : lds_b1, K);
    }
    // fragments + WMMA from As[cur], Bs[cur]
    const unsigned short* Asc = As[cur];
    const unsigned short* Bsc = Bs[cur];
    union { u32x4 u[2]; v16bf v; } fa[4], fb[2];
#pragma unroll
    for (int m = 0; m < 4; ++m) {
      const u32x4* p = (const u32x4*)(Asc + (waveR * 64 + m * 16 + l16) * PAD);
      fa[m].u[0] = p[hh];
      fa[m].u[1] = p[2 + hh];
    }
#pragma unroll
    for (int n = 0; n < 2; ++n) {
      const u32x4* p = (const u32x4*)(Bsc + (waveC * 32 + n * 16 + l16) * PAD);
      fb[n].u[0] = p[hh];
      fb[n].u[1] = p[2 + hh];
    }
#pragma unroll
    for (int m = 0; m < 4; ++m)
#pragma unroll
      for (int n = 0; n < 2; ++n)
        acc[m][n] = __builtin_amdgcn_wmma_f32_16x16x32_bf16(
            false, fa[m].v, false, fb[n].v, (short)0, acc[m][n], false, false);
  }

  // epilogue — C/D layout: VGPR i, half hh: row = i + 8*hh, col = l16
#pragma unroll
  for (int m = 0; m < 4; ++m) {
#pragma unroll
    for (int n = 0; n < 2; ++n) {
      const int col = colBase + waveC * 32 + n * 16 + l16;
      const float bval = bias[(size_t)e * N + col];
#pragma unroll
      for (int i = 0; i < 8; ++i) {
        const int row = rowBase + waveR * 64 + m * 16 + i + 8 * hh;
        float v = acc[m][n][i] + bval;
        if (MODE == 0) {
          actOut[(size_t)row * N + col] = f2bf(gelu_new(v));
        } else {
          const int t = slot_map[row];
          if (t >= 0) dOut[(size_t)t * N + col] = mp[t] * v;
        }
      }
    }
  }
}

extern "C" void kernel_launch(void* const* d_in, const int* in_sizes, int n_in,
                              void* d_out, int out_size, void* d_ws, size_t ws_size,
                              hipStream_t stream) {
  (void)in_sizes; (void)n_in; (void)out_size; (void)ws_size;
  const float* x      = (const float*)d_in[0];
  const float* W_r    = (const float*)d_in[1];
  const float* b_r    = (const float*)d_in[2];
  const float* W_fc   = (const float*)d_in[3];
  const float* b_fc   = (const float*)d_in[4];
  const float* W_proj = (const float*)d_in[5];
  const float* b_proj = (const float*)d_in[6];
  float* out = (float*)d_out;

  // workspace carve (~288 MB):
  //   slot_map | eidx | mp | bf16 buf | bf16 act | bf16 Wt_fc | bf16 Wt_proj
  char* ws = (char*)d_ws;
  size_t off = 0;
  auto carve = [&](size_t bytes) -> char* {
    char* p = ws + off;
    off = (off + bytes + 255) & ~(size_t)255;
    return p;
  };
  int*            slot_map = (int*)carve((size_t)kRows * 4);
  int*            eidx     = (int*)carve((size_t)kTok * 4);
  float*          mp       = (float*)carve((size_t)kTok * 4);
  unsigned short* buf      = (unsigned short*)carve((size_t)kRows * kH * 2);
  unsigned short* act      = (unsigned short*)carve((size_t)kRows * kI * 2);
  unsigned short* wtfc     = (unsigned short*)carve((size_t)kE * kH * kI * 2);
  unsigned short* wtproj   = (unsigned short*)carve((size_t)kE * kI * kH * 2);

  // one-shot weight convert+transpose (amortized over 275 GFLOP of GEMM)
  convert_transpose_kernel<kH, kI><<<dim3(kH / 64, kI / 64, kE), 256, 0, stream>>>(
      W_fc, wtfc);
  convert_transpose_kernel<kI, kH><<<dim3(kI / 64, kH / 64, kE), 256, 0, stream>>>(
      W_proj, wtproj);

  router_kernel<<<kTok / 8, 256, 0, stream>>>(x, W_r, b_r, eidx, mp);
  init_slotmap<<<(kRows + 255) / 256, 256, 0, stream>>>(slot_map);
  priority_kernel<<<kB, 32, 0, stream>>>(eidx, slot_map);
  dispatch_kernel<<<kRows, 128, 0, stream>>>(x, slot_map, buf);
  fallback_kernel<<<kTok, 256, 0, stream>>>(x, mp, out);
  gemm_kernel<kI, kH, 0><<<dim3(kI / 128, kRows / 128), 256, 0, stream>>>(
      buf, wtfc, b_fc, act, nullptr, nullptr, nullptr);
  gemm_kernel<kH, kI, 1><<<dim3(kH / 128, kRows / 128), 256, 0, stream>>>(
      act, wtproj, b_proj, nullptr, out, slot_map, mp);
}
